// FastFFlayer_14851996910183
// MI455X (gfx1250) — matrix operand order repaired and checked by reading, not compiled, not run on hardware
//
#include <hip/hip_runtime.h>
#include <hip/hip_bf16.h>

// ---------------------------------------------------------------------------
// FastFF / DeltaNet forward for MI455X (gfx1250, wave32, WMMA).
//
// Shapes: SLEN=2048, BSZ=8, IN_DIM=1024, H=16, D=64, NPROJ = H*(3D+1) = 3088.
// M = SLEN*BSZ = 16384 tokens.
//
// Roofline: projections = 137 GFLOP (bf16 WMMA, compute-trivial on MI455X);
// total HBM traffic ~500 MB => ~25 us at 23.3 TB/s; the exact sequential
// delta-rule scan (2048 dependent steps x 128 chains) dominates latency.
//
// Pipeline:
//   1) k_layernorm     : x fp32 -> h bf16                  [M,1024]
//   2) k_transpose_cvt : w_slow -> wslow_t bf16 [3088,1024]; w_out -> wout_t
//   3) k_gemm<0>       : raw = h @ w_slow  (double-buffered WMMA, bf16 out)
//   4) k_activation    : elu+1 / sum-norm / sigmoid -> qkv bf16 [M,H,192], beta
//   5) k_recurrence    : exact sequential delta rule, W[64][64] in VGPRs
//   6) k_gemm<1>       : y = x + attn @ w_out  (WMMA, fp32 out + residual)
// ---------------------------------------------------------------------------

#define SLEN   2048
#define BSZ    8
#define IN_DIM 1024
#define NHEAD  16
#define DHEAD  64
#define NPROJ  3088          // 16 * 193
#define MTOK   (SLEN * BSZ)  // 16384

typedef __attribute__((ext_vector_type(16))) __bf16 v16bf;
typedef __attribute__((ext_vector_type(8)))  float  v8f;

union FragBF { v16bf v; uint4 q[2]; };

// ---------------- LayerNorm: fp32 row -> bf16 row ----------------
__global__ void k_layernorm(const float* __restrict__ x,
                            const float* __restrict__ w,
                            const float* __restrict__ b,
                            __hip_bfloat16* __restrict__ out) {
  const int row = blockIdx.x;                    // 0..MTOK-1
  const float* xr = x + (size_t)row * IN_DIM;
  float vals[4];
  float s = 0.f, s2 = 0.f;
#pragma unroll
  for (int i = 0; i < 4; ++i) {
    float v = xr[threadIdx.x + i * 256];
    vals[i] = v; s += v; s2 += v * v;
  }
#pragma unroll
  for (int o = 16; o > 0; o >>= 1) {             // wave32 reduce
    s  += __shfl_down(s,  o, 32);
    s2 += __shfl_down(s2, o, 32);
  }
  __shared__ float sh[2][8];
  __shared__ float mb[2];
  const int wv = threadIdx.x >> 5, ln = threadIdx.x & 31;
  if (ln == 0) { sh[0][wv] = s; sh[1][wv] = s2; }
  __syncthreads();
  if (threadIdx.x == 0) {
    float t = 0.f, t2 = 0.f;
#pragma unroll
    for (int i = 0; i < 8; ++i) { t += sh[0][i]; t2 += sh[1][i]; }
    float mu  = t  / (float)IN_DIM;
    float var = t2 / (float)IN_DIM - mu * mu;
    mb[0] = mu; mb[1] = rsqrtf(var + 1e-5f);
  }
  __syncthreads();
  const float mu = mb[0], rs = mb[1];
  __hip_bfloat16* orow = out + (size_t)row * IN_DIM;
#pragma unroll
  for (int i = 0; i < 4; ++i) {
    int c = threadIdx.x + i * 256;
    orow[c] = __float2bfloat16((vals[i] - mu) * rs * w[c] + b[c]);
  }
}

// -------- Transpose + fp32->bf16 convert: in[R,C] -> out[C,R] --------
__global__ void k_transpose_cvt(const float* __restrict__ in,
                                __hip_bfloat16* __restrict__ out,
                                int R, int C) {
  __shared__ float tile[32][33];
  const int c0 = blockIdx.x * 32, r0 = blockIdx.y * 32;
  const int tx = threadIdx.x, ty = threadIdx.y;   // 32 x 8
#pragma unroll
  for (int j = 0; j < 32; j += 8) {
    int r = r0 + ty + j, c = c0 + tx;
    tile[ty + j][tx] = (r < R && c < C) ? in[(size_t)r * C + c] : 0.f;
  }
  __syncthreads();
#pragma unroll
  for (int j = 0; j < 32; j += 8) {
    int r = r0 + tx, c = c0 + ty + j;
    if (r < R && c < C)
      out[(size_t)c * R + r] = __float2bfloat16(tile[tx][ty + j]);
  }
}

// ---------------- WMMA fragment loaders (ISA 7.12.2 layouts) ----------------
// A (16-bit, 16x32): lane row = lane&15; hi = lane>>4.
//   elements 0..7  = K[k0 + hi*8 .. +7], elements 8..15 = K[k0+16+hi*8 .. +7]
__device__ inline v16bf load_a_frag(const __hip_bfloat16* A, int lda,
                                    int row, int k0, int hi) {
  FragBF f;
  const __hip_bfloat16* p = A + (size_t)row * lda + k0 + hi * 8;
  f.q[0] = *reinterpret_cast<const uint4*>(p);
  f.q[1] = *reinterpret_cast<const uint4*>(p + 16);
  return f.v;
}
// B (32x16): lane col = lane&15; hi = lane>>4.
//   elements 0..7  = K[k0 + hi*8 .. +7], elements 8..15 = K[k0+16+hi*8 .. +7]
// Bt is [N,K] row-major, so two 16B runs per lane.
__device__ inline v16bf load_b_frag(const __hip_bfloat16* Bt, int ldb,
                                    int col, int k0, int hi) {
  FragBF f;
  const __hip_bfloat16* p = Bt + (size_t)col * ldb + k0 + hi * 8;
  f.q[0] = *reinterpret_cast<const uint4*>(p);
  f.q[1] = *reinterpret_cast<const uint4*>(p + 16);
  return f.v;
}

// ---------------- Register-tiled, double-buffered WMMA GEMM ----------------
// C[M,N] = A[M,K](bf16,row) x Bt[N,K](bf16,row).  Block = 256 thr = 8 waves.
// Block tile 128(M) x 64(N); each wave: 16x64 (A-fragment reused x4).
// K-loop is software-pipelined: fragments for k+32 are issued before the
// WMMAs for k, so s_wait_loadcnt becomes partial instead of a full drain.
// MODE 0: store bf16.  MODE 1: store fp32 with residual add.
// No LDS staging: wslow_t/wout_t (6.3/2 MB) and the 32 MB A panel are
// resident in MI455X's 192 MB L2, so redundant B reads across waves hit L2.
template <int MODE>
__global__ void k_gemm(const __hip_bfloat16* __restrict__ A,
                       const __hip_bfloat16* __restrict__ Bt,
                       int M, int N, int K,
                       __hip_bfloat16* __restrict__ Cb,
                       const float* __restrict__ resid,
                       float* __restrict__ Cf) {
  const int lane  = threadIdx.x & 31;
  const int wave  = threadIdx.x >> 5;
  const int hi    = lane >> 4;
  const int row16 = lane & 15;
  const int m0   = blockIdx.x * 128 + wave * 16;
  const int n0   = blockIdx.y * 64;
  const int mrow = m0 + row16;

  int colc[4];
#pragma unroll
  for (int t = 0; t < 4; ++t) {
    int c = n0 + t * 16 + row16;
    colc[t] = c < N ? c : N - 1;                  // clamp edge-tile loads
  }

  v8f acc[4] = {v8f{0}, v8f{0}, v8f{0}, v8f{0}};

  // Prologue: fragments for k0 = 0
  v16bf af = load_a_frag(A, K, mrow, 0, hi);
  v16bf bfr[4];
#pragma unroll
  for (int t = 0; t < 4; ++t) bfr[t] = load_b_frag(Bt, K, colc[t], 0, hi);

  for (int k0 = 0; k0 < K; k0 += 32) {
    // Prefetch next K-slice (branchless: last iter harmlessly re-reads k=0)
    const int kn = (k0 + 32 < K) ? (k0 + 32) : 0;
    v16bf af_n = load_a_frag(A, K, mrow, kn, hi);
    v16bf bf_n[4];
#pragma unroll
    for (int t = 0; t < 4; ++t) bf_n[t] = load_b_frag(Bt, K, colc[t], kn, hi);

#pragma unroll
    for (int t = 0; t < 4; ++t) {
      acc[t] = __builtin_amdgcn_wmma_f32_16x16x32_bf16(
          /*neg_a=*/false, af, /*neg_b=*/false, bfr[t],
          /*c_mod=*/(short)0, acc[t], /*reuse_a=*/false, /*reuse_b=*/false);
    }
    af = af_n;
#pragma unroll
    for (int t = 0; t < 4; ++t) bfr[t] = bf_n[t];
  }

  // C/D layout: VGPR r -> row m0 + r + hi*8 ; col n0 + t*16 + (lane&15)
#pragma unroll
  for (int t = 0; t < 4; ++t) {
#pragma unroll
    for (int r = 0; r < 8; ++r) {
      int rr = m0 + r + hi * 8;
      int cc = n0 + t * 16 + (lane & 15);
      if (cc < N) {
        size_t idx = (size_t)rr * N + cc;
        if (MODE == 0) Cb[idx] = __float2bfloat16(acc[t][r]);
        else           Cf[idx] = resid[idx] + acc[t][r];
      }
    }
  }
}

// ---------------- Activations: elu+1, sum-norm, sigmoid ----------------
__device__ inline float elu_p1(float x) { return x > 0.f ? x + 1.f : __expf(x); }

// One wave per (token m, head h). lane handles d = lane and d = lane+32.
__global__ void k_activation(const __hip_bfloat16* __restrict__ raw,
                             __hip_bfloat16* __restrict__ act,
                             float* __restrict__ betaOut) {
  const int gw   = (blockIdx.x * blockDim.x + threadIdx.x) >> 5;
  const int lane = threadIdx.x & 31;
  const int m = gw >> 4, h = gw & 15;
  const __hip_bfloat16* rp = raw + (size_t)m * NPROJ + h * 193;

  float q0 = elu_p1(__bfloat162float(rp[lane]));
  float q1 = elu_p1(__bfloat162float(rp[lane + 32]));
  float k0 = elu_p1(__bfloat162float(rp[64 + lane]));
  float k1 = elu_p1(__bfloat162float(rp[64 + lane + 32]));
  float v0 = __bfloat162float(rp[128 + lane]);
  float v1 = __bfloat162float(rp[128 + lane + 32]);

  float qs = q0 + q1, ks = k0 + k1;
#pragma unroll
  for (int o = 16; o > 0; o >>= 1) {
    qs += __shfl_xor(qs, o, 32);
    ks += __shfl_xor(ks, o, 32);
  }
  const float rq = 1.f / qs, rk = 1.f / ks;

  __hip_bfloat16* ap = act + ((size_t)m * NHEAD + h) * 192;
  ap[lane]            = __float2bfloat16(q0 * rq);
  ap[lane + 32]       = __float2bfloat16(q1 * rq);
  ap[64 + lane]       = __float2bfloat16(k0 * rk);
  ap[64 + lane + 32]  = __float2bfloat16(k1 * rk);
  ap[128 + lane]      = __float2bfloat16(v0);
  ap[128 + lane + 32] = __float2bfloat16(v1);
  if (lane == 0) {
    float bx = __bfloat162float(rp[192]);
    betaOut[(size_t)m * NHEAD + h] = 1.f / (1.f + __expf(-bx));
  }
}

// ---------------- Exact sequential delta-rule scan ----------------
// 128 blocks = (b,h) chains, 64 threads (2 waves). Thread i owns row i of the
// 64x64 fp32 state W in VGPRs; k/q/v broadcast via LDS (conflict-free, read
// as float4 -> ds_load_b128). qkv for t+2 is prefetched (global_prefetch_b8).
__global__ void k_recurrence(const __hip_bfloat16* __restrict__ qkv,
                             const float* __restrict__ beta,
                             __hip_bfloat16* __restrict__ attn) {
  const int b = blockIdx.x & (BSZ - 1);
  const int h = blockIdx.x >> 3;
  const int i = threadIdx.x;                  // row of W, 0..63

  float W[DHEAD];
#pragma unroll
  for (int j = 0; j < DHEAD; ++j) W[j] = 0.f;

  __shared__ __align__(16) float qs[DHEAD], ks[DHEAD], vs[DHEAD];
  __shared__ float bet;

  const size_t chain_off = (size_t)h * 192;
  const size_t tok_stride = (size_t)NHEAD * 192;  // bf16 elems per token row

  for (int s = 0; s < SLEN; ++s) {
    const size_t tok  = (size_t)s * BSZ + b;
    const size_t base = tok * tok_stride + chain_off;
    qs[i] = __bfloat162float(qkv[base + i]);
    ks[i] = __bfloat162float(qkv[base + 64 + i]);
    vs[i] = __bfloat162float(qkv[base + 128 + i]);
    if (i == 0) bet = beta[tok * NHEAD + h];
    // Prefetch the qkv block two timesteps ahead (384 B = 3 cachelines)
    if (s + 2 < SLEN && i < 3) {
      const __hip_bfloat16* pf =
          qkv + ((size_t)(s + 2) * BSZ + b) * tok_stride + chain_off + i * 64;
      __builtin_prefetch(pf, 0, 1);
    }
    __syncthreads();

    float vold = 0.f;
#pragma unroll
    for (int j4 = 0; j4 < DHEAD; j4 += 4) {
      float4 kk = *reinterpret_cast<const float4*>(&ks[j4]);  // ds_load_b128
      vold += W[j4] * kk.x + W[j4 + 1] * kk.y +
              W[j4 + 2] * kk.z + W[j4 + 3] * kk.w;
    }
    const float delta = bet * (vs[i] - vold);
    float out = 0.f;
#pragma unroll
    for (int j4 = 0; j4 < DHEAD; j4 += 4) {
      float4 kk = *reinterpret_cast<const float4*>(&ks[j4]);
      float4 qq = *reinterpret_cast<const float4*>(&qs[j4]);
      W[j4]     += delta * kk.x;  out += W[j4]     * qq.x;  // W_new q
      W[j4 + 1] += delta * kk.y;  out += W[j4 + 1] * qq.y;
      W[j4 + 2] += delta * kk.z;  out += W[j4 + 2] * qq.z;
      W[j4 + 3] += delta * kk.w;  out += W[j4 + 3] * qq.w;
    }
    attn[tok * (NHEAD * DHEAD) + h * DHEAD + i] = __float2bfloat16(out);
    __syncthreads();
  }
}

// ---------------------------------------------------------------------------
extern "C" void kernel_launch(void* const* d_in, const int* in_sizes, int n_in,
                              void* d_out, int out_size, void* d_ws, size_t ws_size,
                              hipStream_t stream) {
  const float* x      = (const float*)d_in[0];   // [M, 1024]
  const float* ln_w   = (const float*)d_in[1];   // [1024]
  const float* ln_b   = (const float*)d_in[2];   // [1024]
  const float* w_slow = (const float*)d_in[3];   // [1024, 3088]
  const float* w_out  = (const float*)d_in[4];   // [1024, 1024]
  float* y            = (float*)d_out;           // [M, 1024]

  // Workspace layout (256B aligned). Total ~279 MB.
  char* ws = (char*)d_ws;
  size_t off = 0;
  auto alloc = [&](size_t bytes) { char* p = ws + off; off += (bytes + 255) & ~size_t(255); return p; };
  __hip_bfloat16* h_bf    = (__hip_bfloat16*)alloc((size_t)MTOK * IN_DIM * 2);   // 32 MB
  __hip_bfloat16* wslow_t = (__hip_bfloat16*)alloc((size_t)NPROJ * IN_DIM * 2);  // 6.3 MB
  __hip_bfloat16* wout_t  = (__hip_bfloat16*)alloc((size_t)IN_DIM * IN_DIM * 2); // 2 MB
  __hip_bfloat16* raw     = (__hip_bfloat16*)alloc((size_t)MTOK * NPROJ * 2);    // 101 MB
  __hip_bfloat16* qkv     = (__hip_bfloat16*)alloc((size_t)MTOK * NHEAD * 192 * 2); // 101 MB
  float*          betab   = (float*)alloc((size_t)MTOK * NHEAD * 4);             // 1 MB
  __hip_bfloat16* attn    = (__hip_bfloat16*)alloc((size_t)MTOK * IN_DIM * 2);   // 32 MB
  (void)in_sizes; (void)n_in; (void)out_size; (void)ws_size;

  // 1) LayerNorm -> bf16
  k_layernorm<<<MTOK, 256, 0, stream>>>(x, ln_w, ln_b, h_bf);

  // 2) Weight transpose+convert: [K,N] fp32 -> [N,K] bf16
  k_transpose_cvt<<<dim3((NPROJ + 31) / 32, (IN_DIM + 31) / 32), dim3(32, 8), 0, stream>>>(
      w_slow, wslow_t, IN_DIM, NPROJ);
  k_transpose_cvt<<<dim3((IN_DIM + 31) / 32, (IN_DIM + 31) / 32), dim3(32, 8), 0, stream>>>(
      w_out, wout_t, IN_DIM, IN_DIM);

  // 3) qkvb projection: [M,1024] x [1024,3088]
  k_gemm<0><<<dim3(MTOK / 128, (NPROJ + 63) / 64), 256, 0, stream>>>(
      h_bf, wslow_t, MTOK, NPROJ, IN_DIM, raw, nullptr, nullptr);

  // 4) Activations (one wave per token-head)
  k_activation<<<(MTOK * NHEAD * 32) / 256, 256, 0, stream>>>(raw, qkv, betab);

  // 5) Sequential delta-rule scan: 128 chains
  k_recurrence<<<BSZ * NHEAD, DHEAD, 0, stream>>>(qkv, betab, attn);

  // 6) Output projection + residual: y = x + attn @ w_out
  k_gemm<1><<<dim3(MTOK / 128, IN_DIM / 64), 256, 0, stream>>>(
      attn, wout_t, MTOK, IN_DIM, IN_DIM, nullptr, x, y);
}
